// StructureModule_40896678592843
// MI455X (gfx1250) — matrix operand order repaired and checked by reading, not compiled, not run on hardware
//
#include <hip/hip_runtime.h>
#include <hip/hip_bf16.h>
#include <math.h>

// ---------------------------------------------------------------------------
// AlphaFold Invariant Point Attention on MI455X (gfx1250)
// bf16 WMMA pipeline + TDM tensor_load_to_lds + async global->LDS staging
// ---------------------------------------------------------------------------
typedef __attribute__((ext_vector_type(16))) __bf16 v16bf;
typedef __attribute__((ext_vector_type(8)))  float  v8f;
typedef __attribute__((ext_vector_type(4)))  unsigned v4u;
typedef __attribute__((ext_vector_type(8)))  int      v8i;
typedef __attribute__((ext_vector_type(4)))  int      v4i;

#define NRES  768
#define SDIM  384
#define PDIM  128
#define CDIM  16
#define NHEAD 12
#define NPQ   4
#define NPV   8
#define FEAT  2112            // NHEAD * (PDIM + CDIM + 3*NPV + NPV)

#define W_Lc 0.5773502691896258f     // sqrt(1/3)
#define W_Cc 0.23570226039551584f    // sqrt(2/(9*PQ))
#define INV_SQRT_Cc 0.25f            // C^-0.5

// A-operand (16-bit, 16x32) per-lane element -> K offset (matches B fills):
//  lanes 0-15 : e0..15 -> K 0..15
//  lanes 16-31: e0..15 -> K 16..31
__device__ __forceinline__ int koff_a(int e, int khalf) {
  return khalf * 16 + e;
}
__device__ __forceinline__ __bf16 f2bf(float x) { return (__bf16)x; }

__device__ __forceinline__ v8f wmma_bf16(v16bf a, v16bf b, v8f c) {
  return __builtin_amdgcn_wmma_f32_16x16x32_bf16(false, a, false, b,
                                                 (short)0, c, false, false);
}

// Async-copy `n512` x 512B of contiguous global memory into LDS (per wave:
// 32 lanes x b128 per issue).  Tracked on ASYNCcnt.
__device__ __forceinline__ void async_tile_copy(unsigned lds_base,
                                                const char* gsrc,
                                                int lane, int n512) {
  for (int it = 0; it < n512; ++it) {
    unsigned loff = lds_base + it * 512 + lane * 16;
    const char* g = gsrc + it * 512 + lane * 16;
    asm volatile("global_load_async_to_lds_b128 %0, %1, off"
                 :: "v"(loff), "v"(g) : "memory");
  }
}

__device__ __forceinline__ unsigned rfl(unsigned x) {
  return (unsigned)__builtin_amdgcn_readfirstlane((int)x);
}

// Tensor Data Mover: DMA a 2D tile (rows x rowlen f32 elements, row stride
// `stride_elts`) from global memory into LDS.  D# layout per ISA 8.3/8.4:
// group0 = {count|flags, lds_addr, global_addr lo, global_addr hi | type=2},
// group1 = {data_size, tensor dims, tile dims, dim0 stride}.  Tracked on
// TENSORcnt.  All descriptor words forced wave-uniform into SGPRs.
// This toolchain's builtin is the 6-arg form (g0, g1, g2, g3, g4, cpol).
__device__ __forceinline__ void tdm_load_2d(unsigned lds_addr, const void* gaddr,
                                            unsigned rows, unsigned rowlen,
                                            unsigned stride_elts) {
  unsigned long long ga = (unsigned long long)(size_t)gaddr;
  v4u g0;
  g0[0] = rfl(1u);                                     // count=1, user mode
  g0[1] = rfl(lds_addr);                               // lds_addr (bytes)
  g0[2] = rfl((unsigned)(ga & 0xFFFFFFFFu));           // global_addr[31:0]
  g0[3] = rfl((unsigned)((ga >> 32) & 0x1FFFFFFu) | (2u << 30)); // [56:32]|type=2
  v8i g1;
  g1[0] = (int)rfl(2u << 16);                          // data_size=4B, mask=0
  g1[1] = (int)rfl((rowlen & 0xFFFFu) << 16);          // tensor_dim0 lo
  g1[2] = (int)rfl(((rowlen >> 16) & 0xFFFFu) | ((rows & 0xFFFFu) << 16));
  g1[3] = (int)rfl(((rows >> 16) & 0xFFFFu) | ((rowlen & 0xFFFFu) << 16)); // tile_dim0
  g1[4] = (int)rfl(rows & 0xFFFFu);                    // tile_dim1 (tile_dim2=0)
  g1[5] = (int)rfl(stride_elts);                       // tensor_dim0_stride lo
  g1[6] = 0;
  g1[7] = 0;
  v4i z4 = {};
  v8i z8 = {};
  __builtin_amdgcn_tensor_load_to_lds(g0, g1, z4, z4, z8, 0);
}

// ---------------------------------------------------------------------------
// Generic f32-in / f32-out GEMM, bf16 WMMA compute. One wave = one 16x16 tile.
// M, Ncols multiples of 16; K multiple of 32 (true for all uses here).
// ---------------------------------------------------------------------------
__global__ void gemm_bf16_kernel(const float* __restrict__ A,
                                 const float* __restrict__ B,
                                 const float* __restrict__ bias,
                                 float* __restrict__ C,
                                 int M, int K, int Ncols) {
  int wave  = threadIdx.x >> 5;
  int lane  = threadIdx.x & 31;
  int khalf = lane >> 4, l15 = lane & 15;
  int tiles_m = M >> 4, tiles_n = Ncols >> 4;
  int tile = blockIdx.x * (blockDim.x >> 5) + wave;
  if (tile >= tiles_m * tiles_n) return;       // wave-uniform exit
  int tm = tile / tiles_n, tn = tile % tiles_n;
  int arow = tm * 16 + l15;
  int bcol = tn * 16 + l15;
  v8f acc = {};
  for (int k0 = 0; k0 < K; k0 += 32) {
    v16bf a, b;
#pragma unroll
    for (int e = 0; e < 16; ++e) {
      a[e] = f2bf(A[(size_t)arow * K + k0 + koff_a(e, khalf)]);
      b[e] = f2bf(B[(size_t)(k0 + khalf * 16 + e) * Ncols + bcol]);
    }
    acc = wmma_bf16(a, b, acc);
  }
  float bv = bias ? bias[bcol] : 0.0f;
#pragma unroll
  for (int r = 0; r < 8; ++r)
    C[(size_t)(tm * 16 + r + 8 * khalf) * Ncols + bcol] = acc[r] + bv;
}

// ---------------------------------------------------------------------------
// Apply rigid frames; build extended Q~/K~ (len 32), Vcat (len 48), aq/ak.
// One thread per (i, h).
// ---------------------------------------------------------------------------
__global__ void frames_kernel(const float* __restrict__ Pqkv,
                              const float* __restrict__ Pqkp,
                              const float* __restrict__ Pvp,
                              const float* __restrict__ Rm,
                              const float* __restrict__ tv,
                              const float* __restrict__ gamma,
                              float* __restrict__ Qt, float* __restrict__ Kt,
                              float* __restrict__ Vc,
                              float* __restrict__ aq, float* __restrict__ ak) {
  int idx = blockIdx.x * blockDim.x + threadIdx.x;
  if (idx >= NRES * NHEAD) return;
  int i = idx / NHEAD, h = idx % NHEAD;
  float R[9], t[3];
#pragma unroll
  for (int a = 0; a < 9; ++a) R[a] = Rm[i * 9 + a];
#pragma unroll
  for (int x = 0; x < 3; ++x) t[x] = tv[i * 3 + x];
  float w  = log1pf(__expf(gamma[h]));       // softplus
  float sp = W_Lc * W_Cc * w;                // point logit scale

  float* qrow = Qt + ((size_t)h * NRES + i) * 32;
  float* krow = Kt + ((size_t)h * NRES + i) * 32;
#pragma unroll
  for (int c = 0; c < CDIM; ++c) {
    qrow[c] = Pqkv[(size_t)i * 576 +       c * NHEAD + h] * (W_Lc * INV_SQRT_Cc);
    krow[c] = Pqkv[(size_t)i * 576 + 192 + c * NHEAD + h];
  }
  float sq = 0.f, sk = 0.f;
#pragma unroll
  for (int p = 0; p < NPQ; ++p) {
    float ql[3], kl[3];
#pragma unroll
    for (int x = 0; x < 3; ++x) {
      ql[x] = Pqkp[(size_t)i * 288 +       (p * 3 + x) * NHEAD + h];
      kl[x] = Pqkp[(size_t)i * 288 + 144 + (p * 3 + x) * NHEAD + h];
    }
#pragma unroll
    for (int x = 0; x < 3; ++x) {
      float qg = R[x*3]*ql[0] + R[x*3+1]*ql[1] + R[x*3+2]*ql[2] + t[x];
      float kg = R[x*3]*kl[0] + R[x*3+1]*kl[1] + R[x*3+2]*kl[2] + t[x];
      sq += qg * qg; sk += kg * kg;
      qrow[16 + p * 3 + x] = qg * sp;        // W_L*W_C*w folded into Q~
      krow[16 + p * 3 + x] = kg;
    }
  }
#pragma unroll
  for (int z = 28; z < 32; ++z) { qrow[z] = 0.f; krow[z] = 0.f; }
  aq[i * NHEAD + h] = -0.5f * sp * sq;
  ak[i * NHEAD + h] = -0.5f * sp * sk;

  float* vrow = Vc + ((size_t)h * NRES + i) * 48;
#pragma unroll
  for (int c = 0; c < CDIM; ++c)
    vrow[c] = Pqkv[(size_t)i * 576 + 384 + c * NHEAD + h];
#pragma unroll
  for (int p = 0; p < NPV; ++p) {
    float vl[3];
#pragma unroll
    for (int x = 0; x < 3; ++x) vl[x] = Pvp[(size_t)i * 288 + (p * 3 + x) * NHEAD + h];
#pragma unroll
    for (int x = 0; x < 3; ++x)
      vrow[16 + p * 3 + x] = R[x*3]*vl[0] + R[x*3+1]*vl[1] + R[x*3+2]*vl[2] + t[x];
  }
#pragma unroll
  for (int z = 40; z < 48; ++z) vrow[z] = 0.f;
}

// ---------------------------------------------------------------------------
// logits[h][i][j] = W_L * (pair[i,j,:] . W_b[:,h]).  First pair pass.
// grid (768, 12), block 128.  Each wave DMAs its contiguous 16x128 f32
// pair tile (8 KB) into LDS with ONE tensor_load_to_lds, then 4 K-step WMMAs.
// ---------------------------------------------------------------------------
__global__ void pair_bias_kernel(const float* __restrict__ pair,
                                 const float* __restrict__ Wb,
                                 float* __restrict__ logits) {
  __shared__ float sm[4][16 * PDIM];           // 4 waves x 8 KB
  int wave  = threadIdx.x >> 5;
  int lane  = threadIdx.x & 31;
  int khalf = lane >> 4, l15 = lane & 15;
  int i  = blockIdx.x;
  int j0 = (blockIdx.y * 4 + wave) * 16;
  const float* prow = pair + ((size_t)i * NRES + j0) * PDIM;  // 8 KB contiguous

  // TDM: 16 rows x 128 f32, row stride 128 -> LDS
  tdm_load_2d((unsigned)(size_t)&sm[wave][0], prow, 16, PDIM, PDIM);
  __builtin_amdgcn_s_wait_tensorcnt(0);

  const float* sw = &sm[wave][0];
  v8f acc = {};
  for (int k0 = 0; k0 < PDIM; k0 += 32) {
    v16bf a, b;
#pragma unroll
    for (int e = 0; e < 16; ++e) {
      a[e] = f2bf(sw[l15 * PDIM + k0 + koff_a(e, khalf)]);    // ds_load
      int d = k0 + khalf * 16 + e;
      b[e] = f2bf(l15 < NHEAD ? Wb[d * NHEAD + l15] : 0.0f);
    }
    acc = wmma_bf16(a, b, acc);
  }
  int h = l15;                     // C tile: row = j offset, col = head
  if (h < NHEAD) {
    float* out = logits + ((size_t)h * NRES + i) * NRES + j0;
#pragma unroll
    for (int r = 0; r < 8; ++r) out[r + 8 * khalf] = W_Lc * acc[r];
  }
}

// ---------------------------------------------------------------------------
// logits += Q~ K~^T + aq_i + ak_j.  grid (48, 48, 3), block 128 (4 heads).
// One WMMA per (h, i-tile, j-tile): K = 32 = padded 28-dim contraction.
// ---------------------------------------------------------------------------
__global__ void logits_qk_kernel(const float* __restrict__ Qt,
                                 const float* __restrict__ Kt,
                                 const float* __restrict__ aq,
                                 const float* __restrict__ ak,
                                 float* __restrict__ logits) {
  int wave  = threadIdx.x >> 5;
  int lane  = threadIdx.x & 31;
  int khalf = lane >> 4, l15 = lane & 15;
  int h  = blockIdx.z * 4 + wave;
  int i0 = blockIdx.y * 16, j0 = blockIdx.x * 16;
  const float* qb = Qt + ((size_t)h * NRES + i0 + l15) * 32;
  const float* kb = Kt + ((size_t)h * NRES + j0 + l15) * 32;
  v16bf a, b;
#pragma unroll
  for (int e = 0; e < 16; ++e) {
    a[e] = f2bf(qb[koff_a(e, khalf)]);
    b[e] = f2bf(kb[khalf * 16 + e]);     // B col = lane, K = khalf*16+e
  }
  v8f acc = {};
  acc = wmma_bf16(a, b, acc);
  int j = j0 + l15;
  float akj = ak[j * NHEAD + h];
#pragma unroll
  for (int r = 0; r < 8; ++r) {
    int i = i0 + r + 8 * khalf;
    float* p = logits + ((size_t)h * NRES + i) * NRES + j;
    *p += acc[r] + aq[i * NHEAD + h] + akj;
  }
}

// ---------------------------------------------------------------------------
// Row softmax over j, in place.  One wave32 per (h,i) row.
// ---------------------------------------------------------------------------
__global__ void softmax_kernel(float* __restrict__ logits) {
  int wave = threadIdx.x >> 5;
  int lane = threadIdx.x & 31;
  int row = blockIdx.x * (blockDim.x >> 5) + wave;
  if (row >= NHEAD * NRES) return;
  float* p = logits + (size_t)row * NRES;
  float m = -1e30f;
  for (int j = lane; j < NRES; j += 32) m = fmaxf(m, p[j]);
#pragma unroll
  for (int off = 16; off; off >>= 1) m = fmaxf(m, __shfl_xor(m, off, 32));
  float s = 0.f;
  for (int j = lane; j < NRES; j += 32) s += __expf(p[j] - m);
#pragma unroll
  for (int off = 16; off; off >>= 1) s += __shfl_xor(s, off, 32);
  float inv = 1.0f / s;
  for (int j = lane; j < NRES; j += 32) p[j] = __expf(p[j] - m) * inv;
}

// ---------------------------------------------------------------------------
// o_pair[i,h,:] = sum_j attn[h,i,j] * pair[i,j,:].  Second pair pass:
// double-buffered async global->LDS staging of the 32x128 f32 pair slab
// (16 KB, contiguous) per j-chunk; async engine prefetches chunk c+1 while
// WMMA consumes chunk c from LDS.  2 waves/block (one residue i each),
// 64 KB LDS/block.  Writes straight into feats[:, 0:1536].
// ---------------------------------------------------------------------------
#define OPW 2
__global__ void opair_kernel(const float* __restrict__ attn,
                             const float* __restrict__ pair,
                             float* __restrict__ feats) {
  __shared__ float sm[OPW][2][32 * PDIM];      // 2 waves x 2 bufs x 16 KB
  int wave  = threadIdx.x >> 5;
  int lane  = threadIdx.x & 31;
  int khalf = lane >> 4, l15 = lane & 15;
  int i = blockIdx.x * OPW + wave;
  const char* gbase = (const char*)(pair + (size_t)i * NRES * PDIM);

  // prologue: stage chunk 0
  async_tile_copy((unsigned)(size_t)&sm[wave][0][0], gbase, lane, 32);

  v8f acc[8];
#pragma unroll
  for (int dt = 0; dt < 8; ++dt) acc[dt] = (v8f){};

  const int NCH = NRES / 32;
  for (int jc = 0; jc < NCH; ++jc) {
    int cur = jc & 1;
    if (jc + 1 < NCH) {
      // prefetch next chunk into the other buffer, then wait for current
      async_tile_copy((unsigned)(size_t)&sm[wave][cur ^ 1][0],
                      gbase + (size_t)(jc + 1) * 32 * PDIM * 4, lane, 32);
      asm volatile("s_wait_asynccnt 0x20" ::: "memory");   // <=32 left: cur done
    } else {
      asm volatile("s_wait_asynccnt 0x0" ::: "memory");
    }
    int j0 = jc * 32;
    const float* sb = &sm[wave][cur][0];
    v16bf a;                                 // A: rows = heads, K = j
#pragma unroll
    for (int e = 0; e < 16; ++e) {
      int j = j0 + koff_a(e, khalf);
      a[e] = f2bf(l15 < NHEAD ? attn[((size_t)l15 * NRES + i) * NRES + j] : 0.0f);
    }
#pragma unroll
    for (int dt = 0; dt < 8; ++dt) {
      v16bf b;                               // B from LDS: K = j, cols = d
#pragma unroll
      for (int e = 0; e < 16; ++e)
        b[e] = f2bf(sb[(khalf * 16 + e) * PDIM + dt * 16 + l15]);
      acc[dt] = wmma_bf16(a, b, acc[dt]);
    }
  }
#pragma unroll
  for (int dt = 0; dt < 8; ++dt)
#pragma unroll
    for (int r = 0; r < 8; ++r) {
      int h = r + 8 * khalf;
      if (h < NHEAD)
        feats[(size_t)i * FEAT + h * PDIM + dt * 16 + l15] = acc[dt][r];
    }
}

// ---------------------------------------------------------------------------
// o / og: per-head WMMA of attn tile against Vcat [v ; vg] (48 cols).
// grid 48, block 384 (wave = head).  o -> feats[:,1536:1728], og -> ws.
// ---------------------------------------------------------------------------
__global__ void ocat_kernel(const float* __restrict__ attn,
                            const float* __restrict__ Vc,
                            float* __restrict__ feats,
                            float* __restrict__ ogw) {
  int h     = threadIdx.x >> 5;
  int lane  = threadIdx.x & 31;
  int khalf = lane >> 4, l15 = lane & 15;
  int i0 = blockIdx.x * 16;
  v8f acc[3];
#pragma unroll
  for (int nt = 0; nt < 3; ++nt) acc[nt] = (v8f){};
  for (int j0 = 0; j0 < NRES; j0 += 32) {
    if (j0 + 32 < NRES)
      __builtin_prefetch(&attn[((size_t)h * NRES + i0 + l15) * NRES + j0 + 32], 0, 0);
    v16bf a;
#pragma unroll
    for (int e = 0; e < 16; ++e)
      a[e] = f2bf(attn[((size_t)h * NRES + i0 + l15) * NRES + j0 + koff_a(e, khalf)]);
#pragma unroll
    for (int nt = 0; nt < 3; ++nt) {
      v16bf b;
#pragma unroll
      for (int e = 0; e < 16; ++e)
        b[e] = f2bf(Vc[((size_t)h * NRES + j0 + khalf * 16 + e) * 48 + nt * 16 + l15]);
      acc[nt] = wmma_bf16(a, b, acc[nt]);
    }
  }
#pragma unroll
  for (int nt = 0; nt < 3; ++nt)
#pragma unroll
    for (int r = 0; r < 8; ++r) {
      int i   = i0 + r + 8 * khalf;
      int col = nt * 16 + l15;
      if (col < 16)
        feats[(size_t)i * FEAT + 1536 + h * 16 + col] = acc[nt][r];
      else if (col < 40)
        ogw[((size_t)i * NHEAD + h) * 24 + (col - 16)] = acc[nt][r];
    }
}

// ---------------------------------------------------------------------------
// Back to local frame: ol = R^T (og - t); norms; fill feats[:,1728:2112].
// One thread per (i, h, p).
// ---------------------------------------------------------------------------
__global__ void finish_kernel(const float* __restrict__ ogw,
                              const float* __restrict__ Rm,
                              const float* __restrict__ tv,
                              float* __restrict__ feats) {
  int idx = blockIdx.x * blockDim.x + threadIdx.x;
  if (idx >= NRES * NHEAD * NPV) return;
  int p = idx % NPV;
  int h = (idx / NPV) % NHEAD;
  int i = idx / (NPV * NHEAD);
  float og[3];
#pragma unroll
  for (int x = 0; x < 3; ++x)
    og[x] = ogw[((size_t)i * NHEAD + h) * 24 + p * 3 + x] - tv[i * 3 + x];
  float ol[3];
#pragma unroll
  for (int x = 0; x < 3; ++x)      // ol_x = sum_y R[y][x] * og_y
    ol[x] = Rm[i*9 + x] * og[0] + Rm[i*9 + 3 + x] * og[1] + Rm[i*9 + 6 + x] * og[2];
  float n2 = ol[0]*ol[0] + ol[1]*ol[1] + ol[2]*ol[2] + 1e-8f;
#pragma unroll
  for (int x = 0; x < 3; ++x)
    feats[(size_t)i * FEAT + 1728 + h * 24 + p * 3 + x] = ol[x];
  feats[(size_t)i * FEAT + 2016 + h * NPV + p] = sqrtf(n2);
}

// ---------------------------------------------------------------------------
extern "C" void kernel_launch(void* const* d_in, const int* in_sizes, int n_in,
                              void* d_out, int out_size, void* d_ws, size_t ws_size,
                              hipStream_t stream) {
  const float* single = (const float*)d_in[0];
  const float* pair   = (const float*)d_in[1];
  const float* Rm     = (const float*)d_in[2];
  const float* tv     = (const float*)d_in[3];
  const float* W_qkv  = (const float*)d_in[4];
  const float* W_b    = (const float*)d_in[5];
  const float* W_qkp  = (const float*)d_in[6];
  const float* W_vp   = (const float*)d_in[7];
  const float* gamma  = (const float*)d_in[8];
  const float* W_out  = (const float*)d_in[9];
  const float* b_out  = (const float*)d_in[10];
  float* out = (float*)d_out;

  float* p = (float*)d_ws;
  float* Pqkv = p; p += (size_t)NRES * 576;
  float* Pqkp = p; p += (size_t)NRES * 288;
  float* Pvp  = p; p += (size_t)NRES * 288;
  float* Qt   = p; p += (size_t)NHEAD * NRES * 32;
  float* Kt   = p; p += (size_t)NHEAD * NRES * 32;
  float* Vc   = p; p += (size_t)NHEAD * NRES * 48;
  float* aq   = p; p += (size_t)NRES * NHEAD;
  float* ak   = p; p += (size_t)NRES * NHEAD;
  float* ogw  = p; p += (size_t)NRES * NHEAD * 24;
  float* feats= p; p += (size_t)NRES * FEAT;
  float* attn = p; p += (size_t)NHEAD * NRES * NRES;   // logits then attn, in place

  auto gemm = [&](const float* A, const float* B, const float* bias, float* C,
                  int M, int K, int Nc) {
    int tiles  = (M >> 4) * (Nc >> 4);
    int blocks = (tiles + 7) / 8;
    gemm_bf16_kernel<<<dim3(blocks), dim3(256), 0, stream>>>(A, B, bias, C, M, K, Nc);
  };

  // 1) projections
  gemm(single, W_qkv, nullptr, Pqkv, NRES, SDIM, 576);
  gemm(single, W_qkp, nullptr, Pqkp, NRES, SDIM, 288);
  gemm(single, W_vp,  nullptr, Pvp,  NRES, SDIM, 288);

  // 2) frames / extended q,k,v
  frames_kernel<<<dim3((NRES * NHEAD + 127) / 128), dim3(128), 0, stream>>>(
      Pqkv, Pqkp, Pvp, Rm, tv, gamma, Qt, Kt, Vc, aq, ak);

  // 3) pair bias (pair pass 1, TDM-staged)
  pair_bias_kernel<<<dim3(NRES, NHEAD), dim3(128), 0, stream>>>(pair, W_b, attn);

  // 4) + Q~K~ + aq + ak
  logits_qk_kernel<<<dim3(NRES / 16, NRES / 16, 3), dim3(128), 0, stream>>>(
      Qt, Kt, aq, ak, attn);

  // 5) softmax over j
  softmax_kernel<<<dim3((NHEAD * NRES) / 8), dim3(256), 0, stream>>>(attn);

  // 6) o_pair (pair pass 2, double-buffered async pipeline)
  opair_kernel<<<dim3(NRES / OPW), dim3(OPW * 32), 0, stream>>>(attn, pair, feats);

  // 7) o and og
  ocat_kernel<<<dim3(NRES / 16), dim3(384), 0, stream>>>(attn, Vc, feats, ogw);

  // 8) local frame + norms
  finish_kernel<<<dim3((NRES * NHEAD * NPV + 255) / 256), dim3(256), 0, stream>>>(
      ogw, Rm, tv, feats);

  // 9) output projection
  gemm(feats, W_out, b_out, out, NRES, FEAT, SDIM);
}
